// PriorNetwork_25056839205622
// MI455X (gfx1250) — matrix operand order repaired and checked by reading, not compiled, not run on hardware
//
#include <hip/hip_runtime.h>
#include <math.h>

// ---------------------------------------------------------------------------
// Types for CDNA5 WMMA (wave32): v16bf A/B fragments, v8f C/D fragments.
// ---------------------------------------------------------------------------
typedef __attribute__((ext_vector_type(16))) __bf16        v16bf;
typedef __attribute__((ext_vector_type(8)))  float         v8f;
typedef __attribute__((ext_vector_type(4)))  unsigned int  u32x4;
typedef __attribute__((ext_vector_type(4)))  unsigned short u16x4;

union FragAB { v16bf v; u32x4 q[2]; };
union FragC  { v8f   v; float f[8]; };

__device__ __forceinline__ unsigned short f2bf(float f) {       // RNE f32->bf16
  unsigned u = __float_as_uint(f);
  u += 0x7FFFu + ((u >> 16) & 1u);
  return (unsigned short)(u >> 16);
}
__device__ __forceinline__ float bf2f(unsigned short s) {
  return __uint_as_float(((unsigned)s) << 16);
}
__device__ __forceinline__ v8f wmma_bf16(v16bf a, v16bf b, v8f c) {
  // (neg_a, A, neg_b, B, c_mod, C, reuse_a, reuse_b)
  return __builtin_amdgcn_wmma_f32_16x16x32_bf16(false, a, false, b, (short)0, c,
                                                 false, false);
}

// ---------------------------------------------------------------------------
// Split fp32 -> bf16 hi + bf16 lo (x ~= hi + lo, rel err ~2^-16)
// ---------------------------------------------------------------------------
__global__ void k_cvt_split(const float* __restrict__ s,
                            unsigned short* __restrict__ H,
                            unsigned short* __restrict__ L, int n4) {
  int i = blockIdx.x * blockDim.x + threadIdx.x;
  if (i >= n4) return;
  const float4 x = ((const float4*)s)[i];
  u16x4 h, l;
  h[0] = f2bf(x.x); l[0] = f2bf(x.x - bf2f(h[0]));
  h[1] = f2bf(x.y); l[1] = f2bf(x.y - bf2f(h[1]));
  h[2] = f2bf(x.z); l[2] = f2bf(x.z - bf2f(h[2]));
  h[3] = f2bf(x.w); l[3] = f2bf(x.w - bf2f(h[3]));
  ((u16x4*)H)[i] = h;
  ((u16x4*)L)[i] = l;
}

__global__ void k_init_amin(unsigned long long* __restrict__ amin) {
  int i = blockIdx.x * blockDim.x + threadIdx.x;
  if (i < 512) amin[i] = ~0ull;
}

// ---------------------------------------------------------------------------
// Nearest-neighbor argmin:  dist(b,n) = ||c_n||^2 - 2*x_b.c_n  (||x_b||^2 const)
// Block: 256 thr / 8 waves, tile = 128 codes x 128 candidates.
// Codebook staged fp32->split-bf16 in LDS in K-chunks of 64 (norms in fp32).
// Dot via 3x v_wmma_f32_16x16x32_bf16 (split-precision), argmin via shfl +
// packed global_atomic_min_u64.
// ---------------------------------------------------------------------------
__global__ __launch_bounds__(256) void k_nn(
    const unsigned short* __restrict__ xH, const unsigned short* __restrict__ xL,
    const float* __restrict__ cb, unsigned long long* __restrict__ amin, int N) {
  __shared__ unsigned short sBH[128 * 72];   // 64 K-elems + 8 pad per row
  __shared__ unsigned short sBL[128 * 72];
  __shared__ float sN[128];

  const int tid  = threadIdx.x;
  const int lane = tid & 31;
  const int wv   = tid >> 5;
  const int nm   = lane & 15;          // A-row / B-col within fragment
  const int hh   = lane >> 4;          // lane half
  const long long candBase = (long long)blockIdx.y * 128;
  const int rowBase = blockIdx.x * 128 + wv * 16;

  if (tid < 128) sN[tid] = 0.f;
  __syncthreads();

  FragC acc[8];
#pragma unroll
  for (int t = 0; t < 8; ++t)
#pragma unroll
    for (int j = 0; j < 8; ++j) acc[t].f[j] = 0.f;

  const unsigned short* xrH = xH + (long long)(rowBase + nm) * 256;
  const unsigned short* xrL = xL + (long long)(rowBase + nm) * 256;

  for (int k0 = 0; k0 < 256; k0 += 64) {
    // stage 128 cands x 64 K (fp32 -> bf16 hi/lo), accumulate exact fp32 norms
#pragma unroll
    for (int it = 0; it < 8; ++it) {
      int f4  = tid + 256 * it;         // float4 index in 128x64 chunk
      int row = f4 >> 4;
      int c4  = f4 & 15;
      long long g = candBase + row;
      u16x4 hv, lv;
      if (g < (long long)N) {
        const float4 x = *(const float4*)(cb + g * 256 + k0 + c4 * 4);
        atomicAdd(&sN[row], x.x * x.x + x.y * x.y + x.z * x.z + x.w * x.w);
        hv[0] = f2bf(x.x); lv[0] = f2bf(x.x - bf2f(hv[0]));
        hv[1] = f2bf(x.y); lv[1] = f2bf(x.y - bf2f(hv[1]));
        hv[2] = f2bf(x.z); lv[2] = f2bf(x.z - bf2f(hv[2]));
        hv[3] = f2bf(x.w); lv[3] = f2bf(x.w - bf2f(hv[3]));
        if (g + 128 < (long long)N)     // stream-prefetch next cand tile
          __builtin_prefetch(cb + (g + 128) * 256 + k0 + c4 * 4, 0, 1);
      } else {
        hv[0] = hv[1] = hv[2] = hv[3] = 0;
        lv[0] = lv[1] = lv[2] = lv[3] = 0;
      }
      *(u16x4*)&sBH[row * 72 + c4 * 4] = hv;
      *(u16x4*)&sBL[row * 72 + c4 * 4] = lv;
    }
    __syncthreads();

#pragma unroll
    for (int ks = 0; ks < 64; ks += 32) {
      const int k = k0 + ks;
      FragAB aH, aL;
      aH.q[0] = *(const u32x4*)(xrH + k + 8 * hh);
      aH.q[1] = *(const u32x4*)(xrH + k + 16 + 8 * hh);
      aL.q[0] = *(const u32x4*)(xrL + k + 8 * hh);
      aL.q[1] = *(const u32x4*)(xrL + k + 16 + 8 * hh);
#pragma unroll
      for (int t = 0; t < 8; ++t) {
        FragAB bH, bL;
        const unsigned short* pH = &sBH[(t * 16 + nm) * 72 + ks + 8 * hh];
        const unsigned short* pL = &sBL[(t * 16 + nm) * 72 + ks + 8 * hh];
        bH.q[0] = *(const u32x4*)pH; bH.q[1] = *(const u32x4*)(pH + 16);
        bL.q[0] = *(const u32x4*)pL; bL.q[1] = *(const u32x4*)(pL + 16);
        acc[t].v = wmma_bf16(aH.v, bH.v, acc[t].v);
        acc[t].v = wmma_bf16(aL.v, bH.v, acc[t].v);
        acc[t].v = wmma_bf16(aH.v, bL.v, acc[t].v);
      }
    }
    __syncthreads();
  }

  if (tid < 128 && candBase + tid >= (long long)N) sN[tid] = __builtin_inff();
  __syncthreads();

  // per-row argmin over this block's 128 candidates -> global atomic min
#pragma unroll
  for (int r = 0; r < 8; ++r) {
    float bd = __builtin_inff();
    unsigned bi = 0xFFFFFFFFu;
#pragma unroll
    for (int t = 0; t < 8; ++t) {
      int nl = t * 16 + nm;
      float d = sN[nl] - 2.0f * acc[t].f[r];
      unsigned gi = (unsigned)(candBase + nl);
      if (d < bd || (d == bd && gi < bi)) { bd = d; bi = gi; }
    }
#pragma unroll
    for (int off = 1; off <= 8; off <<= 1) {   // reduce within 16-lane half
      float    od = __shfl_xor(bd, off, 32);
      unsigned oi = __shfl_xor(bi, off, 32);
      if (od < bd || (od == bd && oi < bi)) { bd = od; bi = oi; }
    }
    if (nm == 0) {
      int grow = rowBase + r + 8 * hh;        // code row for this half
      unsigned key = __float_as_uint(bd);
      key = (key & 0x80000000u) ? ~key : (key | 0x80000000u); // order-preserving
      unsigned long long pk = ((unsigned long long)key << 32) | (unsigned long long)bi;
      atomicMin(&amin[grow], pk);
    }
  }
}

// gather prev = codebook[argmin], split to bf16 hi/lo
__global__ void k_gather(const float* __restrict__ cb,
                         const unsigned long long* __restrict__ amin,
                         unsigned short* __restrict__ x0H,
                         unsigned short* __restrict__ x0L) {
  int b = blockIdx.x, t = threadIdx.x;   // 64 threads, 4 floats each
  unsigned idx = (unsigned)(amin[b] & 0xFFFFFFFFull);
  const float4 v = *(const float4*)(cb + (long long)idx * 256 + t * 4);
  u16x4 h, l;
  h[0] = f2bf(v.x); l[0] = f2bf(v.x - bf2f(h[0]));
  h[1] = f2bf(v.y); l[1] = f2bf(v.y - bf2f(h[1]));
  h[2] = f2bf(v.z); l[2] = f2bf(v.z - bf2f(h[2]));
  h[3] = f2bf(v.w); l[3] = f2bf(v.w - bf2f(h[3]));
  *(u16x4*)&x0H[b * 256 + t * 4] = h;
  *(u16x4*)&x0L[b * 256 + t * 4] = l;
}

// ---------------------------------------------------------------------------
// Split-bf16 GEMM:  Y = act(X @ W^T + b), X (512 x K), W (Cout x K).
// Block 256 thr / 8 waves: tile 128 rows x 64 cols (4 col-tiles per wave).
// act==1: tanh, write split-bf16; act==0: write fp32.
// ---------------------------------------------------------------------------
__global__ __launch_bounds__(256) void k_gemm(
    const unsigned short* __restrict__ XH, const unsigned short* __restrict__ XL,
    const unsigned short* __restrict__ WH, const unsigned short* __restrict__ WL,
    const float* __restrict__ bias, void* __restrict__ Y,
    unsigned short* __restrict__ YL, int K, int Cout, int act) {
  const int tid  = threadIdx.x;
  const int lane = tid & 31;
  const int wv   = tid >> 5;
  const int nm   = lane & 15;
  const int hh   = lane >> 4;
  const int rowBase = blockIdx.x * 128 + wv * 16;
  const int colBase = blockIdx.y * 64;

  const unsigned short* xrH = XH + (long long)(rowBase + nm) * K;
  const unsigned short* xrL = XL + (long long)(rowBase + nm) * K;

  FragC acc[4];
#pragma unroll
  for (int t = 0; t < 4; ++t)
#pragma unroll
    for (int j = 0; j < 8; ++j) acc[t].f[j] = 0.f;

  for (int k = 0; k < K; k += 32) {
    FragAB aH, aL;
    aH.q[0] = *(const u32x4*)(xrH + k + 8 * hh);
    aH.q[1] = *(const u32x4*)(xrH + k + 16 + 8 * hh);
    aL.q[0] = *(const u32x4*)(xrL + k + 8 * hh);
    aL.q[1] = *(const u32x4*)(xrL + k + 16 + 8 * hh);
#pragma unroll
    for (int t = 0; t < 4; ++t) {
      const long long wr = (long long)(colBase + t * 16 + nm) * K + k + 8 * hh;
      FragAB bH, bL;
      bH.q[0] = *(const u32x4*)(WH + wr); bH.q[1] = *(const u32x4*)(WH + wr + 16);
      bL.q[0] = *(const u32x4*)(WL + wr); bL.q[1] = *(const u32x4*)(WL + wr + 16);
      acc[t].v = wmma_bf16(aH.v, bH.v, acc[t].v);
      acc[t].v = wmma_bf16(aL.v, bH.v, acc[t].v);
      acc[t].v = wmma_bf16(aH.v, bL.v, acc[t].v);
    }
  }

#pragma unroll
  for (int t = 0; t < 4; ++t) {
    const int col = colBase + t * 16 + nm;
    const float bv = bias[col];
#pragma unroll
    for (int r = 0; r < 8; ++r) {
      const int row = rowBase + r + 8 * hh;
      const float y = acc[t].f[r] + bv;
      const long long o = (long long)row * Cout + col;
      if (act) {
        const float ty = tanhf(y);
        const unsigned short th = f2bf(ty);
        ((unsigned short*)Y)[o] = th;
        YL[o] = f2bf(ty - bf2f(th));
      } else {
        ((float*)Y)[o] = y;
      }
    }
  }
}

// elementwise: z = a + b (split-bf16 in/out)
__global__ void k_add2(const unsigned short* __restrict__ aH, const unsigned short* __restrict__ aL,
                       const unsigned short* __restrict__ bH, const unsigned short* __restrict__ bL,
                       unsigned short* __restrict__ zH, unsigned short* __restrict__ zL, int n) {
  int i = blockIdx.x * blockDim.x + threadIdx.x;
  if (i >= n) return;
  float v = bf2f(aH[i]) + bf2f(aL[i]) + bf2f(bH[i]) + bf2f(bL[i]);
  unsigned short h = f2bf(v);
  zH[i] = h; zL[i] = f2bf(v - bf2f(h));
}
// z = a + b + c
__global__ void k_add3(const unsigned short* __restrict__ aH, const unsigned short* __restrict__ aL,
                       const unsigned short* __restrict__ bH, const unsigned short* __restrict__ bL,
                       const unsigned short* __restrict__ cH, const unsigned short* __restrict__ cL,
                       unsigned short* __restrict__ zH, unsigned short* __restrict__ zL, int n) {
  int i = blockIdx.x * blockDim.x + threadIdx.x;
  if (i >= n) return;
  float v = bf2f(aH[i]) + bf2f(aL[i]) + bf2f(bH[i]) + bf2f(bL[i]) + bf2f(cH[i]) + bf2f(cL[i]);
  unsigned short h = f2bf(v);
  zH[i] = h; zL[i] = f2bf(v - bf2f(h));
}

// ---------------------------------------------------------------------------
// Host orchestration
// ---------------------------------------------------------------------------
extern "C" void kernel_launch(void* const* d_in, const int* in_sizes, int n_in,
                              void* d_out, int out_size, void* d_ws, size_t ws_size,
                              hipStream_t stream) {
  const float* codes = (const float*)d_in[0];
  const float* cb    = (const float*)d_in[1];
  const float* Wf[10];
  const float* Bf[10];
  for (int i = 0; i < 10; ++i) { Wf[i] = (const float*)d_in[2 + 2 * i]; Bf[i] = (const float*)d_in[3 + 2 * i]; }
  const int N = in_sizes[1] / 256;           // 500000

  static const long long WSZ[10] = {1024LL * 256, 1048576, 1048576, 1048576, 1048576,
                                    1048576, 1048576, 1048576, 256LL * 1024, 256LL * 1024};

  unsigned char* base = (unsigned char*)d_ws;
  size_t off = 0;
  auto alloc = [&](size_t bytes) -> void* {
    off = (off + 255) & ~(size_t)255;
    void* p = base + off;
    off += bytes;
    return p;
  };
  unsigned long long* amin = (unsigned long long*)alloc(512 * 8);
  unsigned short* codesH = (unsigned short*)alloc(512 * 256 * 2);
  unsigned short* codesL = (unsigned short*)alloc(512 * 256 * 2);
  unsigned short *WH[10], *WL[10];
  for (int i = 0; i < 10; ++i) {
    WH[i] = (unsigned short*)alloc(WSZ[i] * 2);
    WL[i] = (unsigned short*)alloc(WSZ[i] * 2);
  }
  const int AN = 512 * 1024;
  unsigned short* x0H = (unsigned short*)alloc(512 * 256 * 2);
  unsigned short* x0L = (unsigned short*)alloc(512 * 256 * 2);
  unsigned short *iH = (unsigned short*)alloc(AN * 2), *iL = (unsigned short*)alloc(AN * 2);
  unsigned short *h1H = (unsigned short*)alloc(AN * 2), *h1L = (unsigned short*)alloc(AN * 2);
  unsigned short *s2H = (unsigned short*)alloc(AN * 2), *s2L = (unsigned short*)alloc(AN * 2);
  unsigned short *s3H = (unsigned short*)alloc(AN * 2), *s3L = (unsigned short*)alloc(AN * 2);
  unsigned short *t1H = (unsigned short*)alloc(AN * 2), *t1L = (unsigned short*)alloc(AN * 2);
  unsigned short *h2H = (unsigned short*)alloc(AN * 2), *h2L = (unsigned short*)alloc(AN * 2);
  unsigned short *o1H = (unsigned short*)alloc(AN * 2), *o1L = (unsigned short*)alloc(AN * 2);
  unsigned short *o2H = (unsigned short*)alloc(AN * 2), *o2L = (unsigned short*)alloc(AN * 2);
  unsigned short *t2H = (unsigned short*)alloc(AN * 2), *t2L = (unsigned short*)alloc(AN * 2);
  unsigned short *o3H = (unsigned short*)alloc(AN * 2), *o3L = (unsigned short*)alloc(AN * 2);
  unsigned short *otH = (unsigned short*)alloc(AN * 2), *otL = (unsigned short*)alloc(AN * 2);

  auto cvt = [&](const float* s, unsigned short* H, unsigned short* L, long long n) {
    int n4 = (int)(n / 4);
    k_cvt_split<<<(n4 + 255) / 256, 256, 0, stream>>>(s, H, L, n4);
  };
  cvt(codes, codesH, codesL, 512LL * 256);
  for (int i = 0; i < 10; ++i) cvt(Wf[i], WH[i], WL[i], WSZ[i]);

  k_init_amin<<<2, 256, 0, stream>>>(amin);
  dim3 gnn(4, (unsigned)((N + 127) / 128));
  k_nn<<<gnn, 256, 0, stream>>>(codesH, codesL, cb, amin, N);
  k_gather<<<512, 64, 0, stream>>>(cb, amin, x0H, x0L);

  auto gemm = [&](const unsigned short* XH, const unsigned short* XL, int wi,
                  void* Y, unsigned short* YL, int K, int Cout, int act) {
    dim3 g(4, (unsigned)(Cout / 64));
    k_gemm<<<g, 256, 0, stream>>>(XH, XL, WH[wi], WL[wi], Bf[wi], Y, YL, K, Cout, act);
  };

  gemm(x0H, x0L, 0, iH, iL, 256, 1024, 1);                       // i  = tanh(x0 @ w_in^T)
  gemm(iH, iL, 1, h1H, h1L, 1024, 1024, 1);                      // h1
  gemm(h1H, h1L, 2, s2H, s2L, 1024, 1024, 1);                    // s2
  gemm(h1H, h1L, 3, s3H, s3L, 1024, 1024, 1);                    // s3
  k_add2<<<2048, 256, 0, stream>>>(h1H, h1L, s2H, s2L, t1H, t1L, AN);
  gemm(t1H, t1L, 4, h2H, h2L, 1024, 1024, 1);                    // h2 = tanh((h1+s2)@w_h2^T)
  gemm(h1H, h1L, 5, o1H, o1L, 1024, 1024, 1);                    // o1
  gemm(h2H, h2L, 6, o2H, o2L, 1024, 1024, 1);                    // o2
  k_add2<<<2048, 256, 0, stream>>>(h2H, h2L, s3H, s3L, t2H, t2L, AN);
  gemm(t2H, t2L, 7, o3H, o3L, 1024, 1024, 1);                    // o3
  k_add3<<<2048, 256, 0, stream>>>(o1H, o1L, o2H, o2L, o3H, o3L, otH, otL, AN);
  gemm(otH, otL, 8, (float*)d_out, nullptr, 1024, 256, 0);        // mu
  gemm(otH, otL, 9, (float*)d_out + 512 * 256, nullptr, 1024, 256, 0); // logstd
}
